// CausalSelfAttention_47081431498897
// MI455X (gfx1250) — matrix operand order repaired and checked by reading, not compiled, not run on hardware
//
#include <hip/hip_runtime.h>
#include <hip/hip_bf16.h>
#include <stdint.h>

// ---------------------------------------------------------------------------
// CausalSelfAttention forward for MI455X (gfx1250)
//   bf16 WMMA everywhere, flash attention, TDM (tensor_load_to_lds) staging,
//   software-pipelined LDS reads.
// ---------------------------------------------------------------------------

typedef __bf16 bf16_t;
typedef __bf16 v8bf  __attribute__((ext_vector_type(8)));
typedef __bf16 v16bf __attribute__((ext_vector_type(16)));
typedef float  v8f   __attribute__((ext_vector_type(8)));
typedef unsigned int v4u __attribute__((ext_vector_type(4)));
typedef unsigned int v8u __attribute__((ext_vector_type(8)));

union AFrag { v16bf v; v8bf h[2]; };

__device__ __forceinline__ v8f wmma_bf16(v16bf a, v16bf b, v8f c) {
  // D = A(16x32 bf16) * B(32x16 bf16) + C(16x16 f32)
  return __builtin_amdgcn_wmma_f32_16x16x32_bf16(false, a, false, b,
                                                 (short)0, c, false, false);
}

#define BB 4
#define TT 2048
#define CC 1024
#define HH 16
#define DD 64
#define MM (BB * TT)   // 8192 rows
#define N3 (3 * CC)    // 3072
#define KC 128         // K-chunk staged per TDM transfer

// ---------------------------------------------------------------------------
// TDM: issue an async 2-D tile load (tileH x tileW bf16) from a row-major
// matrix (rowLen elements per row) into LDS at byte offset ldsOff.
// D# packing per CDNA5 ISA 8.3 (group0) / 8.4 (group1); 2-D => groups 2/3 NULL.
// ---------------------------------------------------------------------------
__device__ __forceinline__ void tdm_load_2d(const bf16_t* gbase, unsigned ldsOff,
                                            unsigned rowLen, unsigned numRows,
                                            unsigned k0, unsigned n0,
                                            unsigned tileW, unsigned tileH)
{
  unsigned long long ga =
      (unsigned long long)(uintptr_t)(gbase + (size_t)k0 * rowLen + n0);
  v4u g0;
  g0[0] = 1u;                                          // count=1, user mode
  g0[1] = ldsOff;                                      // lds_addr (bytes)
  g0[2] = (unsigned)(ga & 0xffffffffu);                // global_addr[31:0]
  g0[3] = (unsigned)((ga >> 32) & 0x01ffffffu)         // global_addr[56:32]
        | (2u << 30);                                  // type=2 ("image")
  unsigned long long st0 = (unsigned long long)rowLen; // dim0 stride (elems)
  v8u g1;
  g1[0] = 1u << 16;                                    // data_size=1 (2 bytes)
  g1[1] = (rowLen & 0xffffu) << 16;                    // tensor_dim0[15:0]
  g1[2] = (rowLen >> 16) | ((numRows & 0xffffu) << 16);// td0[31:16] | td1[15:0]
  g1[3] = (numRows >> 16) | (tileW << 16);             // td1[31:16] | tile_dim0
  g1[4] = tileH;                                       // tile_dim1 (tile_dim2=0)
  g1[5] = (unsigned)(st0 & 0xffffffffu);               // dim0_stride[31:0]
  g1[6] = (unsigned)((st0 >> 32) & 0xffffu);           // dim0_stride[47:32]
  g1[7] = 0u;                                          // dim1_stride (unused, 2D)
  asm volatile("tensor_load_to_lds %0, %1" :: "s"(g0), "s"(g1) : "memory");
}

// ---------------------------------------------------------------------------
// Kernel 1a: fp32 -> bf16 elementwise convert
// ---------------------------------------------------------------------------
__global__ void k_cvt(const float* __restrict__ in, bf16_t* __restrict__ out, int n) {
  int i = blockIdx.x * blockDim.x + threadIdx.x;
  if (i < n) out[i] = (bf16_t)in[i];
}

// ---------------------------------------------------------------------------
// Kernel 1b: transpose+convert  wt[k*rows + n] = w[n*cols + k]
// ---------------------------------------------------------------------------
__global__ void k_transpose(const float* __restrict__ w, bf16_t* __restrict__ wt,
                            int rows, int cols) {
  int i = blockIdx.x * blockDim.x + threadIdx.x;
  if (i < rows * cols) {
    int n = i / cols;
    int k = i - n * cols;
    wt[(size_t)k * rows + n] = (bf16_t)w[(size_t)n * cols + k];
  }
}

// ---------------------------------------------------------------------------
// Shared GEMM core: block = 4 waves, 64(M) x 64(N) tile, K in 128-chunks.
// B panel (128 x 64 bf16, 16KB) is TDM-staged into double-buffered LDS by
// wave 0; all waves consume it with strided 16-bit LDS reads, software-pipelined one
// 32-K step ahead so WMMAs never wait on fresh LDS issues.
// ---------------------------------------------------------------------------
__device__ __forceinline__ void gemm_core(
    const bf16_t* __restrict__ a_base,   // [M][K=CC] bf16, row-major
    const bf16_t* __restrict__ b_base,   // [K=CC][rowLen] bf16, row-major
    unsigned rowLen, int n0, int row_a,  // row_a = this wave's A row (M)
    int wave, int lane, bf16_t* ldsB,    // ldsB = [2][KC*64]
    v8f acc[4])
{
  const int hf = lane >> 4;
  const bf16_t* arow = a_base + (size_t)row_a * CC + (hf ? 8 : 0);
  const unsigned lds0 = (unsigned)(uintptr_t)ldsB;

  if (wave == 0)
    tdm_load_2d(b_base, lds0, rowLen, CC, 0, (unsigned)n0, 64, KC);

  const int nchunk = CC / KC;   // 8
  for (int c = 0; c < nchunk; ++c) {
    const int k0 = c * KC;
    if (wave == 0) {
      if (c + 1 < nchunk) {
        tdm_load_2d(b_base, lds0 + ((c + 1) & 1) * (KC * 64 * 2),
                    rowLen, CC, (unsigned)(k0 + KC), (unsigned)n0, 64, KC);
        __builtin_amdgcn_s_wait_tensorcnt(1);   // chunk c has landed
      } else {
        __builtin_amdgcn_s_wait_tensorcnt(0);
      }
    }
    __syncthreads();                            // B chunk c visible to block

    if (k0 + KC < CC)
      __builtin_prefetch((const void*)(arow + k0 + KC), 0, 1);

    // Batch all A fragments of this chunk (one load clause).
    AFrag a[4];
#pragma unroll
    for (int s = 0; s < 4; ++s) {
      a[s].h[0] = *(const v8bf*)(arow + k0 + s * 32);
      a[s].h[1] = *(const v8bf*)(arow + k0 + s * 32 + 16);
    }

    const bf16_t* Bl = ldsB + (c & 1) * (KC * 64);
    // Pipeline B fragments one 32-K step ahead.
    v16bf b0[4], b1[4];
#pragma unroll
    for (int t = 0; t < 4; ++t)
#pragma unroll
      for (int e = 0; e < 16; ++e)
        b0[t][e] = Bl[(size_t)((e & 7) + 8 * hf + 16 * (e >> 3)) * 64 + t * 16 + (lane & 15)];
#pragma unroll
    for (int s = 0; s < 4; ++s) {
      if (s < 3) {
        const bf16_t* brn = Bl + (size_t)((s + 1) * 32) * 64;
#pragma unroll
        for (int t = 0; t < 4; ++t)
#pragma unroll
          for (int e = 0; e < 16; ++e)
            b1[t][e] = brn[(size_t)((e & 7) + 8 * hf + 16 * (e >> 3)) * 64 + t * 16 + (lane & 15)];
      }
      acc[0] = wmma_bf16(a[s].v, b0[0], acc[0]);
      acc[1] = wmma_bf16(a[s].v, b0[1], acc[1]);
      acc[2] = wmma_bf16(a[s].v, b0[2], acc[2]);
      acc[3] = wmma_bf16(a[s].v, b0[3], acc[3]);
#pragma unroll
      for (int t = 0; t < 4; ++t) b0[t] = b1[t];
    }
    __syncthreads();                            // all reads of chunk c done
  }
}

// ---------------------------------------------------------------------------
// Kernel 2: qkv = x @ qkv_w^T + b ; scatter to q[b,h,t,d], kT[b,h,d,t], v[b,h,t,d]
// grid: (M/64) x (3072/64) = 128*48 blocks of 128 threads
// ---------------------------------------------------------------------------
__global__ __launch_bounds__(128) void k_gemm_qkv(
    const bf16_t* __restrict__ xb, const bf16_t* __restrict__ wt,
    const float* __restrict__ bias,
    bf16_t* __restrict__ qo, bf16_t* __restrict__ kto, bf16_t* __restrict__ vo)
{
  __shared__ bf16_t ldsB[2][KC * 64];   // 32 KB double buffer

  const int lane = threadIdx.x & 31;
  const int wave = threadIdx.x >> 5;
  const int NT   = N3 / 64;             // 48
  const int bm   = blockIdx.x / NT;
  const int bn   = blockIdx.x - bm * NT;
  const int n0   = bn * 64;
  const int nl   = lane & 15;
  const int hf   = lane >> 4;
  const int row_a = bm * 64 + wave * 16 + nl;

  v8f acc[4] = {};
  gemm_core(xb, wt, N3, n0, row_a, wave, lane, &ldsB[0][0], acc);

  // Epilogue: C layout -> VGPR r, lane l : M = r + 8*(l>>4), N = l&15
#pragma unroll
  for (int nt = 0; nt < 4; ++nt) {
    const int col = n0 + nt * 16 + nl;           // [0,3072)
    const int s   = col >> 10;                   // 0=q 1=k 2=v
    const int rem = col & (CC - 1);
    const int h   = rem >> 6;
    const int d   = rem & (DD - 1);
    const float bv = bias[col];
#pragma unroll
    for (int r = 0; r < 8; ++r) {
      const int m = bm * 64 + wave * 16 + r + 8 * hf;
      const int b = m >> 11;                     // /T
      const int t = m & (TT - 1);
      const bf16_t o = (bf16_t)(acc[nt][r] + bv);
      const size_t bh = (size_t)(b * HH + h);
      if (s == 0)      qo[(bh * TT + t) * DD + d] = o;
      else if (s == 1) kto[(bh * DD + d) * TT + t] = o;
      else             vo[(bh * TT + t) * DD + d] = o;
    }
  }
}

// ---------------------------------------------------------------------------
// Kernel 3: flash attention. One wave owns a 16-row q tile; key blocks of 32.
// V fragments are fetched at loop top so their latency hides under softmax.
// ---------------------------------------------------------------------------
__global__ __launch_bounds__(128) void k_attn(
    const bf16_t* __restrict__ qg, const bf16_t* __restrict__ ktg,
    const bf16_t* __restrict__ vg, bf16_t* __restrict__ yb)
{
  __shared__ bf16_t plds[4][16 * 32];   // per-wave P-tile bounce buffer

  const int lane = threadIdx.x & 31;
  const int wave = threadIdx.x >> 5;
  const int wid  = blockIdx.x * 4 + wave;
  const int qt   = wid & 127;            // T/16 = 128 q tiles
  const int bh   = wid >> 7;             // [0, B*H)
  const int t0   = qt << 4;
  const int nl   = lane & 15;
  const int hf   = lane >> 4;

  // Load q A-fragments for K(d) = 0..31 and 32..63
  const bf16_t* qrow = qg + ((size_t)bh * TT + t0 + nl) * DD + (hf ? 8 : 0);
  AFrag aq0, aq1;
  aq0.h[0] = *(const v8bf*)(qrow);
  aq0.h[1] = *(const v8bf*)(qrow + 16);
  aq1.h[0] = *(const v8bf*)(qrow + 32);
  aq1.h[1] = *(const v8bf*)(qrow + 48);

  v8f o[4] = {};
  float mrow[8], lrow[8];
#pragma unroll
  for (int r = 0; r < 8; ++r) { mrow[r] = -1e30f; lrow[r] = 0.0f; }

  bf16_t* pl = plds[wave];
  const bf16_t* ktbase = ktg + (size_t)bh * DD * TT;
  const bf16_t* vbase  = vg + (size_t)bh * TT * DD + nl;        // d = nl (+16*nt)
  const int nkb = (t0 + 16 + 31) >> 5;   // ceil((t0+16)/32) key blocks

  for (int kb = 0; kb < nkb; ++kb) {
    const int j0 = kb << 5;

    // ---- fetch kT fragments (for S) and V fragments (for O) up front
    const bf16_t* kb0 = ktbase + j0 + nl;             // key = j0+nl, elems over d
    const bf16_t* kb1 = kb0 + (size_t)32 * TT;        // d offset 32
    v16bf b00, b01, b10, b11;
#pragma unroll
    for (int e = 0; e < 16; ++e) {
      const size_t kk = (size_t)((e & 7) + 8 * hf + 16 * (e >> 3)) * TT;
      b00[e] = kb0[kk];
      b01[e] = kb0[kk + 16];
      b10[e] = kb1[kk];
      b11[e] = kb1[kk + 16];
    }
    v16bf bv[4];
    {
      const bf16_t* vr = vbase + (size_t)j0 * DD;     // col d = nt*16+nl, elems over keys
#pragma unroll
      for (int nt = 0; nt < 4; ++nt)
#pragma unroll
        for (int e = 0; e < 16; ++e)
          bv[nt][e] = vr[(size_t)((e & 7) + 8 * hf + 16 * (e >> 3)) * DD + nt * 16];
    }
    if (kb + 1 < nkb) {
      __builtin_prefetch((const void*)(kb0 + 32), 0, 1);
      __builtin_prefetch((const void*)(vbase + (size_t)(j0 + 32) * DD), 0, 1);
    }

    // ---- S = (q @ k^T) for 16 rows x 32 keys
    v8f s0 = {}, s1 = {};
    s0 = wmma_bf16(aq0.v, b00, s0);
    s0 = wmma_bf16(aq1.v, b10, s0);
    s1 = wmma_bf16(aq0.v, b01, s1);
    s1 = wmma_bf16(aq1.v, b11, s1);

    // ---- online softmax update (fp32), P tile -> LDS as bf16
    const int col0 = j0 + nl;
#pragma unroll
    for (int r = 0; r < 8; ++r) {
      const int rowi = t0 + r + 8 * hf;
      float v0 = s0[r] * 0.125f;   // 1/sqrt(64)
      float v1 = s1[r] * 0.125f;
      if (col0 > rowi)      v0 = -1e30f;   // causal mask
      if (col0 + 16 > rowi) v1 = -1e30f;
      float tmax = fmaxf(v0, v1);
#pragma unroll
      for (int msk = 1; msk < 16; msk <<= 1)
        tmax = fmaxf(tmax, __shfl_xor(tmax, msk, 32));
      const float mnew = fmaxf(mrow[r], tmax);
      const float corr = __expf(mrow[r] - mnew);
      const float p0 = __expf(v0 - mnew);
      const float p1 = __expf(v1 - mnew);
      float rs = p0 + p1;
#pragma unroll
      for (int msk = 1; msk < 16; msk <<= 1)
        rs += __shfl_xor(rs, msk, 32);
      lrow[r] = lrow[r] * corr + rs;
      mrow[r] = mnew;
#pragma unroll
      for (int nt = 0; nt < 4; ++nt) o[nt][r] *= corr;
      pl[(r + 8 * hf) * 32 + nl]      = (bf16_t)p0;
      pl[(r + 8 * hf) * 32 + nl + 16] = (bf16_t)p1;
    }

    // LDS is in-order within a wave; drain DS before cross-lane re-read
    asm volatile("s_wait_dscnt 0" ::: "memory");

    // Re-read P in A-fragment (16x32) layout
    AFrag pa;
    {
      const bf16_t* pr = pl + nl * 32 + (hf ? 8 : 0);
      pa.h[0] = *(const v8bf*)(pr);
      pa.h[1] = *(const v8bf*)(pr + 16);
    }
    asm volatile("" ::: "memory");

    // ---- O += P @ V ; B rows = keys j0..j0+31 over elements, d over lanes
    o[0] = wmma_bf16(pa.v, bv[0], o[0]);
    o[1] = wmma_bf16(pa.v, bv[1], o[1]);
    o[2] = wmma_bf16(pa.v, bv[2], o[2]);
    o[3] = wmma_bf16(pa.v, bv[3], o[3]);
  }

  // ---- normalize and write y[b, t, h*64 + d] in bf16
  const int b = bh >> 4;          // / H
  const int h = bh & (HH - 1);
#pragma unroll
  for (int nt = 0; nt < 4; ++nt) {
#pragma unroll
    for (int r = 0; r < 8; ++r) {
      const int rowi = t0 + r + 8 * hf;
      const int d = nt * 16 + nl;
      const float val = o[nt][r] / lrow[r];
      yb[((size_t)b * TT + rowi) * CC + h * DD + d] = (bf16_t)val;
    }
  }
}

// ---------------------------------------------------------------------------
// Kernel 4: out = y @ out_w^T + out_b  (fp32 output)
// grid: (M/64) x (1024/64) = 128*16 blocks of 128 threads
// ---------------------------------------------------------------------------
__global__ __launch_bounds__(128) void k_gemm_out(
    const bf16_t* __restrict__ yb, const bf16_t* __restrict__ wt,
    const float* __restrict__ bias, float* __restrict__ out)
{
  __shared__ bf16_t ldsB[2][KC * 64];

  const int lane = threadIdx.x & 31;
  const int wave = threadIdx.x >> 5;
  const int NT   = CC / 64;             // 16
  const int bm   = blockIdx.x / NT;
  const int bn   = blockIdx.x - bm * NT;
  const int n0   = bn * 64;
  const int nl   = lane & 15;
  const int hf   = lane >> 4;
  const int row_a = bm * 64 + wave * 16 + nl;

  v8f acc[4] = {};
  gemm_core(yb, wt, CC, n0, row_a, wave, lane, &ldsB[0][0], acc);

#pragma unroll
  for (int nt = 0; nt < 4; ++nt) {
    const int col = n0 + nt * 16 + nl;
    const float bv = bias[col];
#pragma unroll
    for (int r = 0; r < 8; ++r) {
      const int m = bm * 64 + wave * 16 + r + 8 * hf;
      out[(size_t)m * CC + col] = acc[nt][r] + bv;
    }
  }
}

// ---------------------------------------------------------------------------
// Host launcher
// ---------------------------------------------------------------------------
extern "C" void kernel_launch(void* const* d_in, const int* in_sizes, int n_in,
                              void* d_out, int out_size, void* d_ws, size_t ws_size,
                              hipStream_t stream)
{
  const float* x     = (const float*)d_in[0];   // [B,T,C]
  const float* qkv_w = (const float*)d_in[1];   // [3C,C]
  const float* qkv_b = (const float*)d_in[2];   // [3C]
  const float* out_w = (const float*)d_in[3];   // [C,C]
  const float* out_b = (const float*)d_in[4];   // [C]
  float* out = (float*)d_out;                   // [B,T,C] fp32

  char* ws = (char*)d_ws;
  bf16_t* xb  = (bf16_t*)(ws + 0);          // 16 MB  x in bf16
  bf16_t* qwt = (bf16_t*)(ws + 16777216);   //  6 MB  qkv_w^T bf16 [1024][3072]
  bf16_t* owt = (bf16_t*)(ws + 23068672);   //  2 MB  out_w^T bf16 [1024][1024]
  bf16_t* q   = (bf16_t*)(ws + 25165824);   // 16 MB  q  [B,H,T,D]
  bf16_t* kt  = (bf16_t*)(ws + 41943040);   // 16 MB  kT [B,H,D,T]
  bf16_t* vv  = (bf16_t*)(ws + 58720256);   // 16 MB  v  [B,H,T,D]
  bf16_t* yb  = (bf16_t*)(ws + 75497472);   // 16 MB  attn output [B,T,C] bf16
  // total 88 MB

  k_cvt<<<(MM * CC + 255) / 256, 256, 0, stream>>>(x, xb, MM * CC);
  k_transpose<<<(N3 * CC + 255) / 256, 256, 0, stream>>>(qkv_w, qwt, N3, CC);
  k_transpose<<<(CC * CC + 255) / 256, 256, 0, stream>>>(out_w, owt, CC, CC);

  // (8192/64) * (3072/64) = 128*48 blocks
  k_gemm_qkv<<<6144, 128, 0, stream>>>(xb, qwt, qkv_b, q, kt, vv);

  // B*H*(T/16) = 8192 wave tiles / 4 waves per block
  k_attn<<<2048, 128, 0, stream>>>(q, kt, vv, yb);

  // (8192/64) * (1024/64) = 128*16 blocks
  k_gemm_out<<<2048, 128, 0, stream>>>(yb, owt, out_b, out);
}
